// Attention_85736137163331
// MI455X (gfx1250) — compile-verified, matrix operand order checked by reading, not run on hardware
//
#include <hip/hip_runtime.h>
#include <hip/hip_bf16.h>

#define B_SZ  4
#define S_LEN 4096
#define D_MOD 512

typedef __attribute__((ext_vector_type(16))) __bf16 v16bf;
typedef __attribute__((ext_vector_type(8)))  float  v8f;
typedef int v4i __attribute__((ext_vector_type(4)));

union FragBF {
  v16bf  v;
  uint4  q[2];
  __bf16 e[16];
};

__device__ __forceinline__ v8f wmma_bf16(const FragBF& a, const FragBF& b, v8f c) {
  return __builtin_amdgcn_wmma_f32_16x16x32_bf16(false, a.v, false, b.v, (short)0, c, false, false);
}

// Read a 16(row)x32(K) bf16 fragment (row-major, leading dim ldE elems); works for LDS or global.
// A-layout: lane r=lane&15 holds row r; K runs [h*8,h*8+8) and [16+h*8,16+h*8+8).
__device__ __forceinline__ FragBF frag16x32(const __bf16* base, int row0, int ldE, int kBase, int lane) {
  const int r = lane & 15, h = lane >> 4;
  const __bf16* p = base + (size_t)(row0 + r) * ldE + kBase;
  FragBF f;
  f.q[0] = *(const uint4*)(p + h * 8);
  f.q[1] = *(const uint4*)(p + 16 + h * 8);
  return f;
}

// ---------------- async global -> LDS staging (CDNA5 path, with fallback) ----------------
#if __has_builtin(__builtin_amdgcn_global_load_async_to_lds_b128) && __has_builtin(__builtin_amdgcn_s_wait_asynccnt)
#define HAS_ASYNC_LDS 1
typedef __attribute__((address_space(1))) v4i* gas1_v4i;
typedef __attribute__((address_space(3))) v4i* las3_v4i;
#else
#define HAS_ASYNC_LDS 0
#endif

__device__ __forceinline__ void copy16B(const __bf16* g, __bf16* l) {
#if HAS_ASYNC_LDS
  // generic->integer->AS pointer reinterpret; low 32 bits of a generic LDS address are the LDS offset
  __builtin_amdgcn_global_load_async_to_lds_b128(
      (gas1_v4i)(unsigned long long)g,
      (las3_v4i)(unsigned int)(unsigned long long)l, 0, 0);
#else
  *(uint4*)l = *(const uint4*)g;
#endif
}
__device__ __forceinline__ void async_wait0() {
#if HAS_ASYNC_LDS
  __builtin_amdgcn_s_wait_asynccnt(0);
#endif
}

// ---------------- W f32 -> bf16 pre-conversion (one-time; W is L2-resident afterwards) ----------------
__global__ void __launch_bounds__(256) cvt_w(
    const float* __restrict__ Wq, const float* __restrict__ Wk, const float* __restrict__ Wv,
    __bf16* __restrict__ wq, __bf16* __restrict__ wk, __bf16* __restrict__ wv)
{
  const int z = blockIdx.y;
  const float* src = (z == 0) ? Wq : (z == 1) ? Wk : Wv;
  __bf16* dst = (z == 0) ? wq : (z == 1) ? wk : wv;
  const size_t i4 = ((size_t)blockIdx.x * 256 + threadIdx.x) * 4;   // over 512*512
  const float4 f = *(const float4*)(src + i4);
  union { __bf16 e[4]; uint2 u; } o;
  o.e[0] = (__bf16)f.x; o.e[1] = (__bf16)f.y; o.e[2] = (__bf16)f.z; o.e[3] = (__bf16)f.w;
  *(uint2*)(dst + i4) = o.u;
}

// ---------------- Projection: Q/K row-major bf16, V transposed [b][d][s] bf16 ----------------
__global__ void __launch_bounds__(128) proj_qkv(
    const float* __restrict__ x,
    const __bf16* __restrict__ wq, const __bf16* __restrict__ wk, const __bf16* __restrict__ wv,
    __bf16* __restrict__ qOut, __bf16* __restrict__ kOut, __bf16* __restrict__ vTOut)
{
  const int tid = threadIdx.x, lane = tid & 31, w = tid >> 5;
  const int z = blockIdx.z;
  const __bf16* W = (z == 0) ? wq : (z == 1) ? wk : wv;
  const int row0 = blockIdx.x * 16;                 // over B*S
  const int colBase = blockIdx.y * 256 + w * 64;    // each wave: 4 x 16 output cols
  const int h = lane >> 4, n = lane & 15;

  __shared__ __bf16 xTile[16 * 40];                 // padded ld=40 (bank-safe, 16B aligned runs)

  v8f acc[4] = {};

  for (int kb = 0; kb < 16; ++kb) {
    const int k0 = kb * 32;
    __syncthreads();
    {   // cooperative stage of x 16x32 tile (f32 -> bf16), coalesced
      const int idx = tid * 4, r = idx >> 5, c = idx & 31;
      const float4 fv = *(const float4*)(x + (size_t)(row0 + r) * D_MOD + k0 + c);
      __bf16* dp = xTile + r * 40 + c;
      dp[0] = (__bf16)fv.x; dp[1] = (__bf16)fv.y; dp[2] = (__bf16)fv.z; dp[3] = (__bf16)fv.w;
    }
    __syncthreads();
    const FragBF a = frag16x32(xTile, 0, 40, 0, lane);
    #pragma unroll
    for (int f = 0; f < 4; ++f) {
      // B col e = W row e (out = x * W^T); bf16 W -> two straight b128 loads, no converts
      const FragBF b = frag16x32(W, colBase + f * 16, D_MOD, k0, lane);
      acc[f] = wmma_bf16(a, b, acc[f]);
    }
  }

  #pragma unroll
  for (int f = 0; f < 4; ++f) {
    #pragma unroll
    for (int i = 0; i < 8; ++i) {
      const int m = h * 8 + i;                      // C layout: row = 8*h + vgpr
      const int grow = row0 + m;
      const int col = colBase + f * 16 + n;
      const __bf16 val = (__bf16)acc[f][i];
      if (z == 0)      qOut[(size_t)grow * D_MOD + col] = val;
      else if (z == 1) kOut[(size_t)grow * D_MOD + col] = val;
      else {                                        // V stored transposed per batch
        const int bb = grow >> 12, sI = grow & (S_LEN - 1);
        vTOut[((size_t)bb * D_MOD + col) * S_LEN + sI] = val;
      }
    }
  }
}

// ---------------- Flash attention: 32 queries per WG, 4 waves split D into 128-slices ----------------
#define KT_LD 520      // K/Q tile leading dim (bf16): 1040B rows -> conflict-free, 16B aligned
#define VT_LD 40       // V^T tile leading dim (bf16): 80B rows
#define SP_LD 33       // score-partial leading dim (f32)
#define NEG_BIG (-1.0e30f)

#define ATTN_SMEM (32*KT_LD*2 /*qTile*/ + 32*KT_LD*2 /*kTile*/ + D_MOD*VT_LD*2 /*vTileT*/ \
                 + 4*32*SP_LD*4 /*sPart*/ + 32*VT_LD*2 /*pTile*/ + 64*4 /*alpha+linv*/)

__global__ void __launch_bounds__(128) attn_fwd(
    const __bf16* __restrict__ Q, const __bf16* __restrict__ K, const __bf16* __restrict__ VT,
    const int* __restrict__ pad, float* __restrict__ out)
{
  const int tid = threadIdx.x, lane = tid & 31, w = tid >> 5;
  const int h = lane >> 4, n = lane & 15;
  const int qt = blockIdx.x, batch = blockIdx.y;
  const int q0 = qt * 32;
  const size_t bRow = (size_t)batch * S_LEN;
  const int dwBase = w * 128;                       // this wave's D-slice

  extern __shared__ char smem[];
  __bf16* qTile  = (__bf16*)smem;                      // 32 x KT_LD
  __bf16* kTile  = qTile + 32 * KT_LD;                 // 32 x KT_LD
  __bf16* vTileT = kTile + 32 * KT_LD;                 // D_MOD x VT_LD  ([d][k])
  float*  sPart  = (float*)(vTileT + D_MOD * VT_LD);   // 4 x 32 x SP_LD
  __bf16* pTile  = (__bf16*)(sPart + 4 * 32 * SP_LD);  // 32 x VT_LD
  float*  alphaA = (float*)(pTile + 32 * VT_LD);       // 32
  float*  linvA  = alphaA + 32;                        // 32

  {   // stage Q tile once (32 x 512 bf16) -- async, completion covered by first tile's wait
    const __bf16* src = Q + (bRow + q0) * D_MOD;
    #pragma unroll
    for (int j = 0; j < 16; ++j) {
      const int id = tid + 128 * j, r = id >> 6, c8 = id & 63;
      copy16B(src + (size_t)r * D_MOD + c8 * 8, qTile + r * KT_LD + c8 * 8);
    }
  }

  v8f O[2][8] = {};            // 32q x 128d f32 accumulator per wave
  float mRow = NEG_BIG, lRow = 0.0f;   // running stats (valid in wave 0, row = tid)

  const int nkt = qt + 1;      // causal: key tiles 0..qt
  for (int kt = 0; kt < nkt; ++kt) {
    const int k0 = kt * 32;
    __syncthreads();           // protect LDS tiles from previous iteration's readers
    {   // stage K tile (32 x 512)
      const __bf16* src = K + (bRow + k0) * D_MOD;
      #pragma unroll
      for (int j = 0; j < 16; ++j) {
        const int id = tid + 128 * j, r = id >> 6, c8 = id & 63;
        copy16B(src + (size_t)r * D_MOD + c8 * 8, kTile + r * KT_LD + c8 * 8);
      }
    }
    {   // stage V^T tile (512 x 32), contiguous along s in global
      const __bf16* src = VT + (size_t)batch * D_MOD * S_LEN + k0;
      #pragma unroll
      for (int j = 0; j < 16; ++j) {
        const int id = tid + 128 * j, d = id >> 2, c8 = id & 3;
        copy16B(src + (size_t)d * S_LEN + c8 * 8, vTileT + d * VT_LD + c8 * 8);
      }
    }
    async_wait0();
    __syncthreads();

    // partial scores over this wave's 128-d slice: 2x2 16x16 tiles, K-chained WMMA
    float* sp = sPart + w * 32 * SP_LD;
    #pragma unroll
    for (int rs = 0; rs < 2; ++rs) {
      FragBF qa[4];
      #pragma unroll
      for (int dc = 0; dc < 4; ++dc)
        qa[dc] = frag16x32(qTile, rs * 16, KT_LD, dwBase + dc * 32, lane);
      #pragma unroll
      for (int ks = 0; ks < 2; ++ks) {
        v8f acc = {};
        #pragma unroll
        for (int dc = 0; dc < 4; ++dc) {
          const FragBF kb = frag16x32(kTile, ks * 16, KT_LD, dwBase + dc * 32, lane);
          acc = wmma_bf16(qa[dc], kb, acc);
        }
        #pragma unroll
        for (int i = 0; i < 8; ++i)
          sp[(rs * 16 + h * 8 + i) * SP_LD + ks * 16 + n] = acc[i];
      }
    }
    __syncthreads();

    // online softmax: wave 0, one lane per query row
    if (tid < 32) {
      const int qg = q0 + tid;
      const float scale = 0.04419417382415922f;     // 1/sqrt(512)
      float* s0 = sPart + tid * SP_LD;              // reuse wave-0 region for masked scores
      float mx = NEG_BIG;
      for (int k = 0; k < 32; ++k) {
        float s = 0.f;
        #pragma unroll
        for (int ww = 0; ww < 4; ++ww) s += sPart[ww * 32 * SP_LD + tid * SP_LD + k];
        s *= scale;
        const int kg = k0 + k;
        const bool ok = (kg <= qg) && (pad[bRow + kg] != 0);
        s = ok ? s : NEG_BIG;
        s0[k] = s;
        mx = fmaxf(mx, s);
      }
      const float mNew = fmaxf(mRow, mx);
      const float alpha = __expf(mRow - mNew);      // l==0 makes the degenerate case harmless
      float sum = 0.f;
      for (int k = 0; k < 32; ++k) {
        const float s = s0[k];
        const float p = (s <= NEG_BIG * 0.5f) ? 0.f : __expf(s - mNew);
        sum += p;
        pTile[tid * VT_LD + k] = (__bf16)p;
      }
      lRow = alpha * lRow + sum;
      mRow = mNew;
      alphaA[tid] = alpha;
    }
    __syncthreads();

    // rescale O by alpha and accumulate P*V on this wave's D-slice
    #pragma unroll
    for (int rs = 0; rs < 2; ++rs) {
      float av[8];
      #pragma unroll
      for (int i = 0; i < 8; ++i) av[i] = alphaA[rs * 16 + h * 8 + i];
      const FragBF pa = frag16x32(pTile, rs * 16, VT_LD, 0, lane);  // P 16x32 (K = 32 keys)
      #pragma unroll
      for (int ct = 0; ct < 8; ++ct) {
        #pragma unroll
        for (int i = 0; i < 8; ++i) O[rs][ct][i] *= av[i];
        const FragBF vb = frag16x32(vTileT, dwBase + ct * 16, VT_LD, 0, lane); // B col=d, K=k
        O[rs][ct] = wmma_bf16(pa, vb, O[rs][ct]);
      }
    }
  }

  if (tid < 32) linvA[tid] = (lRow > 0.f) ? (1.f / lRow) : 0.f;
  __syncthreads();

  #pragma unroll
  for (int rs = 0; rs < 2; ++rs) {
    float lv[8];
    #pragma unroll
    for (int i = 0; i < 8; ++i) lv[i] = linvA[rs * 16 + h * 8 + i];
    #pragma unroll
    for (int ct = 0; ct < 8; ++ct) {
      #pragma unroll
      for (int i = 0; i < 8; ++i) {
        const size_t row = bRow + q0 + rs * 16 + h * 8 + i;
        out[row * D_MOD + dwBase + ct * 16 + n] = O[rs][ct][i] * lv[i];
      }
    }
  }
}

extern "C" void kernel_launch(void* const* d_in, const int* in_sizes, int n_in,
                              void* d_out, int out_size, void* d_ws, size_t ws_size,
                              hipStream_t stream) {
  const float* x  = (const float*)d_in[0];
  const int*   pm = (const int*)d_in[1];
  const float* Wq = (const float*)d_in[2];
  const float* Wk = (const float*)d_in[3];
  const float* Wv = (const float*)d_in[4];

  const size_t elems = (size_t)B_SZ * S_LEN * D_MOD;   // 8M
  const size_t wElems = (size_t)D_MOD * D_MOD;         // 256K
  __bf16* qB  = (__bf16*)d_ws;
  __bf16* kB  = qB + elems;
  __bf16* vT  = kB + elems;
  __bf16* wqB = vT + elems;
  __bf16* wkB = wqB + wElems;
  __bf16* wvB = wkB + wElems;                          // ~49.5 MB total workspace

  cvt_w<<<dim3((int)(wElems / (256 * 4)), 3), 256, 0, stream>>>(Wq, Wk, Wv, wqB, wkB, wvB);
  proj_qkv<<<dim3(B_SZ * S_LEN / 16, 2, 3), 128, 0, stream>>>(x, wqB, wkB, wvB, qB, kB, vT);

  (void)hipFuncSetAttribute((const void*)attn_fwd,
                            hipFuncAttributeMaxDynamicSharedMemorySize, ATTN_SMEM);
  attn_fwd<<<dim3(S_LEN / 32, B_SZ), 128, ATTN_SMEM, stream>>>(qB, kB, vT, pm, (float*)d_out);
}
